// GATBase_32332513804856
// MI455X (gfx1250) — matrix operand order repaired and checked by reading, not compiled, not run on hardware
//
#include <hip/hip_runtime.h>
#include <hip/hip_bf16.h>

// ---------------------------------------------------------------------------
// Problem constants (match reference)
// ---------------------------------------------------------------------------
#define NNODES 50000   // 3125 * 16 = 625 * 80 -> no GEMM row tails
#define NEDGES 800000
#define NGRAPH 64
#define F_IN   128
#define HIDF   64
#define NHEADS 4
#define NCLS   10
#define NEG_SLOPE 0.2f

typedef __attribute__((ext_vector_type(16))) __bf16 v16bf;
typedef __attribute__((ext_vector_type(8)))  float  v8f;
typedef unsigned short u16;

union ABf { v16bf v; uint4 q[2]; };

// pack two f32 into one dword of two bf16 (truncating round)
__device__ __forceinline__ unsigned pkbf(float lo, float hi) {
    return (__float_as_uint(lo) >> 16) | (__float_as_uint(hi) & 0xffff0000u);
}

// order-preserving uint encoding of float for atomicMax-based segment max
__device__ __forceinline__ unsigned ford(float f) {
    unsigned u = __float_as_uint(f);
    return (u & 0x80000000u) ? ~u : (u | 0x80000000u);
}
__device__ __forceinline__ float fdec(unsigned e) {
    return (e & 0x80000000u) ? __uint_as_float(e ^ 0x80000000u)
                             : __uint_as_float(~e);
}

__device__ __forceinline__ void get_edge(const int* ei, int E, int e, int& s, int& d) {
    if (e < E) { s = ei[e]; d = ei[E + e]; }
    else       { s = e - E; d = e - E; }        // implicit self-loop
}

// ---------------------------------------------------------------------------
// f32 -> bf16 conversion, 8 elements/thread (one-time per layer; removes all
// conversion work from the GEMM hot loop)
// ---------------------------------------------------------------------------
__global__ void cvt_bf16x8(const float4* __restrict__ in, uint4* __restrict__ out,
                           long long n8)
{
    long long i = (long long)blockIdx.x * blockDim.x + threadIdx.x;
    if (i >= n8) return;
    float4 f0 = in[2 * i], f1 = in[2 * i + 1];
    uint4 o;
    o.x = pkbf(f0.x, f0.y); o.y = pkbf(f0.z, f0.w);
    o.z = pkbf(f1.x, f1.y); o.w = pkbf(f1.z, f1.w);
    out[i] = o;
}

// ---------------------------------------------------------------------------
// WMMA GEMM:  Y[N,M] = Xb[N,K] * Wb[M,K]^T (+ bias[M]), bf16 in / f32 out.
// One wave computes an 80x64 macro-tile: 5 row tiles x 4 col tiles,
// 20 accumulators. Per K-step: B fragments loaded once (reused 5x),
// A fragments reused 4x, 20 back-to-back v_wmma_f32_16x16x32_bf16.
// Fragment packing per CDNA5 ISA wave32 layouts (CDNA5 ISA §7.12.2):
//   A 16x32 bf16 : lane l16 -> row; VGPRs 0-3: K = half*8+0..7,
//                  VGPRs 4-7: K = 16+half*8+0..7   (2 x b128 per fragment)
//   B 32x16 bf16 : lane l16 -> col; VGPRs 0-3: K = half*8+0..7,
//                  VGPRs 4-7: K = 16+half*8+0..7   (2 x b128 per fragment)
//   D 16x16 f32  : lane l16 -> col; VGPR v -> row v + 8*half
// Requires: N % 80 == 0, K % 32 == 0, M % 64 == 0 (true for all layers here).
// ---------------------------------------------------------------------------
__global__ __launch_bounds__(32)
void gemm_wmma_bf16(const u16* __restrict__ Xb, const u16* __restrict__ Wb,
                    const float* __restrict__ bias, float* __restrict__ Y,
                    int N, int K, int M)
{
    const int lane = threadIdx.x & 31;
    const int half = lane >> 4;          // 0: lanes 0-15, 1: lanes 16-31
    const int l16  = lane & 15;
    const int tm0  = blockIdx.x * 4;     // first of 4 column tiles
    const int tn0  = blockIdx.y * 5;     // first of 5 row tiles

    v8f acc[5][4] = {};

    const u16* wp[4];
#pragma unroll
    for (int t = 0; t < 4; ++t)
        wp[t] = Wb + (size_t)((tm0 + t) * 16 + l16) * K;
    const u16* xp[5];
#pragma unroll
    for (int r = 0; r < 5; ++r)
        xp[r] = Xb + (size_t)((tn0 + r) * 16 + l16) * K;

    for (int k0 = 0; k0 < K; k0 += 32) {
        ABf bf[4];
#pragma unroll
        for (int t = 0; t < 4; ++t) {
            const u16* p = wp[t] + k0 + half * 8;
            bf[t].q[0] = *(const uint4*)(p);
            bf[t].q[1] = *(const uint4*)(p + 16);
        }
#pragma unroll
        for (int r = 0; r < 5; ++r) {
            ABf a;
            a.q[0] = *(const uint4*)(xp[r] + k0 + half * 8);
            a.q[1] = *(const uint4*)(xp[r] + k0 + 16 + half * 8);
#pragma unroll
            for (int t = 0; t < 4; ++t)
                acc[r][t] = __builtin_amdgcn_wmma_f32_16x16x32_bf16(
                    false, a.v, false, bf[t].v, (short)0, acc[r][t], false, false);
        }
    }

    // store; lane -> column, VGPR v -> row v + 8*half
#pragma unroll
    for (int r = 0; r < 5; ++r) {
#pragma unroll
        for (int t = 0; t < 4; ++t) {
            const int col = (tm0 + t) * 16 + l16;
            const float bv = bias ? bias[col] : 0.f;
#pragma unroll
            for (int v = 0; v < 8; ++v) {
                int m = (tn0 + r) * 16 + half * 8 + v;
                Y[(size_t)m * M + col] = acc[r][t][v] + bv;
            }
        }
    }
}

// ---------------------------------------------------------------------------
// Attention logits: s[n,h] = <h[n,h,:], a_src[h]>, d[n,h] = <h[n,h,:], a_dst[h]>
// ---------------------------------------------------------------------------
__global__ void sd_kernel(const float* __restrict__ h, const float* __restrict__ asrc,
                          const float* __restrict__ adst, float* __restrict__ s,
                          float* __restrict__ d, int N, int H, int C)
{
    int idx = blockIdx.x * blockDim.x + threadIdx.x;
    if (idx >= N * H) return;
    int n = idx / H, hh = idx % H;
    const float* hp = h + (size_t)n * H * C + (size_t)hh * C;
    const float* as = asrc + hh * C;
    const float* ad = adst + hh * C;
    float ss = 0.f, dd = 0.f;
    for (int c = 0; c < C; ++c) { ss += hp[c] * as[c]; dd += hp[c] * ad[c]; }
    s[idx] = ss; d[idx] = dd;
}

// segment max of leaky-relu edge logits (ordered-uint atomicMax)
__global__ void edge_max_kernel(const int* __restrict__ ei, const float* __restrict__ s,
                                const float* __restrict__ d, unsigned* __restrict__ emax,
                                int E, int N, int H)
{
    int e = blockIdx.x * blockDim.x + threadIdx.x;
    if (e >= E + N) return;
    int sn, dn; get_edge(ei, E, e, sn, dn);
    for (int hh = 0; hh < H; ++hh) {
        float v = s[sn * H + hh] + d[dn * H + hh];
        v = (v > 0.f) ? v : v * NEG_SLOPE;
        atomicMax(&emax[dn * H + hh], ford(v));
    }
}

__global__ void decode_kernel(const unsigned* __restrict__ enc, float* __restrict__ out, int n)
{
    int i = blockIdx.x * blockDim.x + threadIdx.x;
    if (i >= n) return;
    float v = fdec(enc[i]);
    out[i] = (v == v && v > -3.0e38f && v < 3.0e38f) ? v : 0.f;  // where(isfinite, ., 0)
}

// ex = exp(e - emax[dst]); store to alpha buffer; den[dst] += ex
__global__ void edge_ex_kernel(const int* __restrict__ ei, const float* __restrict__ s,
                               const float* __restrict__ d, const float* __restrict__ emax,
                               float* __restrict__ alpha, float* __restrict__ den,
                               int E, int N, int H)
{
    int e = blockIdx.x * blockDim.x + threadIdx.x;
    if (e >= E + N) return;
    int sn, dn; get_edge(ei, E, e, sn, dn);
    for (int hh = 0; hh < H; ++hh) {
        float v = s[sn * H + hh] + d[dn * H + hh];
        v = (v > 0.f) ? v : v * NEG_SLOPE;
        float ex = expf(v - emax[dn * H + hh]);
        alpha[(size_t)e * H + hh] = ex;
        atomicAdd(&den[dn * H + hh], ex);
    }
}

__global__ void edge_alpha_kernel(const int* __restrict__ ei, const float* __restrict__ den,
                                  float* __restrict__ alpha, int E, int N, int H)
{
    long long idx = (long long)blockIdx.x * blockDim.x + threadIdx.x;
    long long total = (long long)(E + N) * H;
    if (idx >= total) return;
    int e = (int)(idx / H), hh = (int)(idx % H);
    int sn, dn; get_edge(ei, E, e, sn, dn); (void)sn;
    alpha[idx] /= (den[dn * H + hh] + 1e-16f);
}

// out[dst, f] += h_t[src, f] * alpha[e, f/C]
__global__ void edge_agg_kernel(const int* __restrict__ ei, const float* __restrict__ alpha,
                                const float* __restrict__ ht, float* __restrict__ out,
                                int E, int N, int H, int C)
{
    long long idx = (long long)blockIdx.x * blockDim.x + threadIdx.x;
    long long total = (long long)(E + N) * H * C;
    long long stride = (long long)gridDim.x * blockDim.x;
    int HC = H * C;
    for (; idx < total; idx += stride) {
        int e = (int)(idx / HC);
        int f = (int)(idx % HC);
        int hh = f / C;
        int sn, dn; get_edge(ei, E, e, sn, dn);
        atomicAdd(&out[(size_t)dn * HC + f],
                  ht[(size_t)sn * HC + f] * alpha[(size_t)e * H + hh]);
    }
}

// x = elu(x + bias[f])
__global__ void bias_elu_kernel(float* __restrict__ x, const float* __restrict__ b,
                                long long total, int F)
{
    long long idx = (long long)blockIdx.x * blockDim.x + threadIdx.x;
    if (idx >= total) return;
    float v = x[idx] + b[(int)(idx % F)];
    x[idx] = (v > 0.f) ? v : (expf(v) - 1.f);
}

// graph pooling: sum, max, counts
__global__ void pool_kernel(const float* __restrict__ h, const int* __restrict__ batch,
                            float* __restrict__ psum, unsigned* __restrict__ pmax,
                            float* __restrict__ counts, int N, int C)
{
    int idx = blockIdx.x * blockDim.x + threadIdx.x;
    if (idx >= N * C) return;
    int n = idx / C, c = idx % C;
    int g = batch[n];
    float v = h[idx];
    atomicAdd(&psum[g * C + c], v);
    atomicMax(&pmax[g * C + c], ford(v));
    if (c == 0) atomicAdd(&counts[g], 1.0f);
}

__global__ void gcat_kernel(const float* __restrict__ psum, const unsigned* __restrict__ pmax,
                            const float* __restrict__ counts, float* __restrict__ g,
                            int B, int C)
{
    int idx = blockIdx.x * blockDim.x + threadIdx.x;
    if (idx >= B * C) return;
    int b = idx / C, c = idx % C;
    float cnt = counts[b];
    g[b * 2 * C + c]     = psum[b * C + c] / fmaxf(cnt, 1.0f);
    g[b * 2 * C + C + c] = (cnt > 0.f) ? fdec(pmax[b * C + c]) : 0.f;
}

// small dense layers: thread per (row, out-feature)
__global__ void fc_kernel(const float* __restrict__ X, const float* __restrict__ W,
                          const float* __restrict__ b, float* __restrict__ Y,
                          int Rows, int K, int M, int relu)
{
    int idx = blockIdx.x * blockDim.x + threadIdx.x;
    if (idx >= Rows * M) return;
    int r = idx / M, m = idx % M;
    const float* xr = X + (size_t)r * K;
    const float* wr = W + (size_t)m * K;
    float acc = b[m];
    for (int k = 0; k < K; ++k) acc += xr[k] * wr[k];
    Y[idx] = relu ? fmaxf(acc, 0.f) : acc;
}

// ---------------------------------------------------------------------------
// Host orchestration
// ---------------------------------------------------------------------------
extern "C" void kernel_launch(void* const* d_in, const int* in_sizes, int n_in,
                              void* d_out, int out_size, void* d_ws, size_t ws_size,
                              hipStream_t stream)
{
    const float* x     = (const float*)d_in[0];
    const int*   ei    = (const int*)  d_in[1];
    const int*   batch = (const int*)  d_in[2];
    const float* W_emb = (const float*)d_in[3];
    const float* b_emb = (const float*)d_in[4];
    const float* W1    = (const float*)d_in[5];
    const float* as1   = (const float*)d_in[6];
    const float* ad1   = (const float*)d_in[7];
    const float* b1    = (const float*)d_in[8];
    const float* W2    = (const float*)d_in[9];
    const float* as2   = (const float*)d_in[10];
    const float* ad2   = (const float*)d_in[11];
    const float* b2    = (const float*)d_in[12];
    const float* W3    = (const float*)d_in[13];
    const float* as3   = (const float*)d_in[14];
    const float* ad3   = (const float*)d_in[15];
    const float* b3    = (const float*)d_in[16];
    const float* Wc1   = (const float*)d_in[17];
    const float* bc1   = (const float*)d_in[18];
    const float* Wc2   = (const float*)d_in[19];
    const float* bc2   = (const float*)d_in[20];

    const int N = NNODES, E = NEDGES, ET = NEDGES + NNODES;
    const int HC = HIDF * NHEADS;   // 256

    // workspace layout
    char* ws = (char*)d_ws;
    size_t off = 0;
    auto alloc = [&](size_t bytes) -> void* {
        void* p = ws + off;
        off = (off + bytes + 255) & ~(size_t)255;
        return p;
    };
    float*    bufA  = (float*)   alloc((size_t)N * HC * 4);
    float*    bufB  = (float*)   alloc((size_t)N * HC * 4);
    float*    bufC  = (float*)   alloc((size_t)N * HC * 4);
    u16*      Xbf   = (u16*)     alloc((size_t)N * HC * 2);   // bf16 activations
    u16*      Wbf   = (u16*)     alloc((size_t)HC * HC * 2);  // bf16 weights
    float*    sbuf  = (float*)   alloc((size_t)N * NHEADS * 4);
    float*    dbuf  = (float*)   alloc((size_t)N * NHEADS * 4);
    unsigned* emaxE = (unsigned*)alloc((size_t)N * NHEADS * 4);
    float*    emaxF = (float*)   alloc((size_t)N * NHEADS * 4);
    float*    den   = (float*)   alloc((size_t)N * NHEADS * 4);
    float*    alpha = (float*)   alloc((size_t)ET * NHEADS * 4);
    float*    counts= (float*)   alloc((size_t)NGRAPH * 4);
    float*    psum  = (float*)   alloc((size_t)NGRAPH * HIDF * 4);
    unsigned* pmax  = (unsigned*)alloc((size_t)NGRAPH * HIDF * 4);
    float*    gbuf  = (float*)   alloc((size_t)NGRAPH * 2 * HIDF * 4);
    float*    c1out = (float*)   alloc((size_t)NGRAPH * HIDF * 4);
    (void)ws_size; (void)n_in; (void)in_sizes; (void)out_size;

    const int TPB = 256;
    auto cdiv = [](long long a, long long b) { return (int)((a + b - 1) / b); };

    // bf16 GEMM with one-time operand conversion: Y = Xf[N,K] @ Wf[M,K]^T (+bias)
    auto gemm = [&](const float* Xf, const float* Wf, const float* bias,
                    float* Y, int K, int M) {
        cvt_bf16x8<<<cdiv((long long)N * K / 8, TPB), TPB, 0, stream>>>(
            (const float4*)Xf, (uint4*)Xbf, (long long)N * K / 8);
        cvt_bf16x8<<<cdiv((long long)M * K / 8, TPB), TPB, 0, stream>>>(
            (const float4*)Wf, (uint4*)Wbf, (long long)M * K / 8);
        gemm_wmma_bf16<<<dim3(M / 64, N / 80), 32, 0, stream>>>(
            Xbf, Wbf, bias, Y, N, K, M);
    };

    // one GAT layer: hin[N,Cin] -> hout[N,H*C] (bias+elu applied in-place)
    auto gat_layer = [&](const float* hin, int Cin, const float* W,
                         const float* asrc, const float* adst, const float* bias,
                         int H, int C, float* ht, float* hout) {
        const int HCL = H * C;
        gemm(hin, W, nullptr, ht, Cin, HCL);
        sd_kernel<<<cdiv((long long)N * H, TPB), TPB, 0, stream>>>(
            ht, asrc, adst, sbuf, dbuf, N, H, C);
        hipMemsetAsync(emaxE, 0, (size_t)N * H * 4, stream);
        hipMemsetAsync(den,   0, (size_t)N * H * 4, stream);
        hipMemsetAsync(hout,  0, (size_t)N * HCL * 4, stream);
        edge_max_kernel<<<cdiv(ET, TPB), TPB, 0, stream>>>(ei, sbuf, dbuf, emaxE, E, N, H);
        decode_kernel<<<cdiv((long long)N * H, TPB), TPB, 0, stream>>>(emaxE, emaxF, N * H);
        edge_ex_kernel<<<cdiv(ET, TPB), TPB, 0, stream>>>(ei, sbuf, dbuf, emaxF, alpha, den, E, N, H);
        edge_alpha_kernel<<<cdiv((long long)ET * H, TPB), TPB, 0, stream>>>(ei, den, alpha, E, N, H);
        edge_agg_kernel<<<cdiv((long long)ET * HCL, TPB), TPB, 0, stream>>>(
            ei, alpha, ht, hout, E, N, H, C);
        bias_elu_kernel<<<cdiv((long long)N * HCL, TPB), TPB, 0, stream>>>(
            hout, bias, (long long)N * HCL, HCL);
    };

    // 1) node embedding: bufA = x @ W_emb^T + b_emb   [N,64]
    gemm(x, W_emb, b_emb, bufA, F_IN, HIDF);

    // 2) GAT stack
    gat_layer(bufA, HIDF, W1, as1, ad1, b1, NHEADS, HIDF, bufB, bufC);  // 64  -> 256
    gat_layer(bufC, HC,   W2, as2, ad2, b2, NHEADS, HIDF, bufA, bufB);  // 256 -> 256
    gat_layer(bufB, HC,   W3, as3, ad3, b3, 1,      HIDF, bufC, bufA);  // 256 -> 64

    // 3) graph pooling (mean | max)
    hipMemsetAsync(counts, 0, (size_t)NGRAPH * 4, stream);
    hipMemsetAsync(psum,   0, (size_t)NGRAPH * HIDF * 4, stream);
    hipMemsetAsync(pmax,   0, (size_t)NGRAPH * HIDF * 4, stream);
    pool_kernel<<<cdiv((long long)N * HIDF, TPB), TPB, 0, stream>>>(
        bufA, batch, psum, pmax, counts, N, HIDF);
    gcat_kernel<<<cdiv((long long)NGRAPH * HIDF, TPB), TPB, 0, stream>>>(
        psum, pmax, counts, gbuf, NGRAPH, HIDF);

    // 4) classifier
    fc_kernel<<<cdiv((long long)NGRAPH * HIDF, TPB), TPB, 0, stream>>>(
        gbuf, Wc1, bc1, c1out, NGRAPH, 2 * HIDF, HIDF, 1);
    fc_kernel<<<cdiv((long long)NGRAPH * NCLS, TPB), TPB, 0, stream>>>(
        c1out, Wc2, bc2, (float*)d_out, NGRAPH, HIDF, NCLS, 0);
}
